// InputFeedRNNDecoder_53403623359155
// MI455X (gfx1250) — matrix operand order repaired and checked by reading, not compiled, not run on hardware
//
#include <hip/hip_runtime.h>
#include <hip/hip_bf16.h>

// ---------------------------------------------------------------------------
// InputFeedRNNDecoder for MI455X (gfx1250, wave32, WMMA).
// Sequential T=50 chain of skinny GEMMs -> bf16 WMMA (f32 accumulate) with
// weights repacked once per launch into a B-tile layout with contiguous k (k = 16*(lane/16) + element).
// K loop is split per A-segment so the hot loop is pure load+wmma+ptr-bump.
// ---------------------------------------------------------------------------

#define HDIM 1024
#define EDIM 512
#define BDIM 32
#define TDIM 50
#define SDIM 100

typedef __attribute__((ext_vector_type(16))) __bf16        v16bf;
typedef __attribute__((ext_vector_type(8)))  float         v8f;
typedef __attribute__((ext_vector_type(8)))  unsigned int  v8u;

__device__ __forceinline__ unsigned short f2bf(float f) {
    unsigned int u = __float_as_uint(f);
    u += 0x7FFFu + ((u >> 16) & 1u);      // round-to-nearest-even
    return (unsigned short)(u >> 16);
}

// ---------------- one-time conversion / repack kernels ----------------------

__global__ void cvt_f32_bf16(const float* __restrict__ src,
                             unsigned short* __restrict__ dst, int n) {
    int i = blockIdx.x * blockDim.x + threadIdx.x;
    if (i < n) dst[i] = f2bf(src[i]);
}

// Pack B = W^T for gates = A @ W^T.  Element (k, n) = srcA[n*lenA+k] for
// k < lenA else srcB[n*lenB + k-lenA].  Tile (32K x 16N): dword layout
// dst[tile*256 + lane*8 + vgpr], lane = n%16 + 16*(k>=16), vgpr=(k%16)/2.
__global__ void repack_nk(unsigned int* __restrict__ dst,
                          const float* __restrict__ srcA, int lenA,
                          const float* __restrict__ srcB, int lenB,
                          int K, int N) {
    int tid = blockIdx.x * blockDim.x + threadIdx.x;
    int totalDwords = (K >> 1) * N;
    if (tid >= totalDwords) return;
    int KT   = K >> 5;
    int tile = tid >> 8;
    int r    = tid & 255;
    int lane = r >> 3;
    int vgpr = r & 7;
    int nt = tile / KT, kt = tile % KT;
    int n  = nt * 16 + (lane & 15);
    int k  = kt * 32 + ((lane >> 4) << 4) + (vgpr << 1);
    float e0, e1;
    if (k < lenA) { e0 = srcA[(long long)n * lenA + k];
                    e1 = srcA[(long long)n * lenA + k + 1]; }
    else          { int kk = k - lenA;
                    e0 = srcB[(long long)n * lenB + kk];
                    e1 = srcB[(long long)n * lenB + kk + 1]; }
    dst[tid] = (unsigned int)f2bf(e0) | ((unsigned int)f2bf(e1) << 16);
}

// Pack B for q = h @ Wa : element (k, n) = Wa[k*N + n]
__global__ void repack_kn(unsigned int* __restrict__ dst,
                          const float* __restrict__ src, int K, int N) {
    int tid = blockIdx.x * blockDim.x + threadIdx.x;
    int totalDwords = (K >> 1) * N;
    if (tid >= totalDwords) return;
    int KT   = K >> 5;
    int tile = tid >> 8;
    int r    = tid & 255;
    int lane = r >> 3;
    int vgpr = r & 7;
    int nt = tile / KT, kt = tile % KT;
    int n  = nt * 16 + (lane & 15);
    int k  = kt * 32 + ((lane >> 4) << 4) + (vgpr << 1);
    float e0 = src[(long long)k * N + n];
    float e1 = src[(long long)(k + 1) * N + n];
    dst[tid] = (unsigned int)f2bf(e0) | ((unsigned int)f2bf(e1) << 16);
}

// ---------------- state init ------------------------------------------------

__global__ void init_state(const float* __restrict__ h0in,
                           const float* __restrict__ c0in,
                           float* __restrict__ h0f, float* __restrict__ h1f,
                           float* __restrict__ c0w, float* __restrict__ c1w,
                           unsigned short* __restrict__ h0b,
                           unsigned short* __restrict__ h1b,
                           unsigned short* __restrict__ feedb) {
    int tid = blockIdx.x * blockDim.x + threadIdx.x;
    if (tid < 2 * BDIM * HDIM) {
        int layer = tid >> 15;
        int idx   = tid & 32767;
        float hv = h0in[tid];
        float cv = c0in[tid];
        if (layer == 0) { h0f[idx] = hv; h0b[idx] = f2bf(hv); c0w[idx] = cv; }
        else            { h1f[idx] = hv; h1b[idx] = f2bf(hv); c1w[idx] = cv; }
    } else if (tid < 2 * BDIM * HDIM + BDIM * HDIM) {
        feedb[tid - 2 * BDIM * HDIM] = 0;   // bf16 zero
    }
}

// ---------------- WMMA GEMM, M = 32 fixed -----------------------------------
// A rows are a concatenation of up to 3 bf16 segments (segment lengths are
// multiples of 32).  K loop split per segment: inner loop is 8x b128 loads,
// 4x v_wmma, pointer bumps.  Each wave: 2 N-tiles x 2 M-tiles.
// mode 0: C[row*ldC+col] = acc
// mode 1: v=tanh(acc); C[row*ldC+col]=v; auxbf[row*ldAux+col]=bf16(v)
__global__ void wmma_gemm32(const unsigned int* __restrict__ Btiles,
                            const unsigned short* __restrict__ A0, int len0, long long sA0,
                            const unsigned short* __restrict__ A1, int len1, long long sA1,
                            const unsigned short* __restrict__ A2, int len2, long long sA2,
                            int K, int N,
                            float* __restrict__ C, long long ldC,
                            int mode,
                            unsigned short* __restrict__ auxbf, long long ldAux) {
    const int lane = threadIdx.x & 31;
    const int wave = (blockIdx.x * blockDim.x + threadIdx.x) >> 5;
    const int ntGroups = N >> 5;                 // 2 N-tiles per wave
    if (wave >= ntGroups) return;
    const int nt0 = wave * 2;
    const int KT  = K >> 5;
    const int row = lane & 15;
    const int hi  = lane >> 4;

    v8f acc00 = {}; v8f acc01 = {}; v8f acc10 = {}; v8f acc11 = {};

    const unsigned short* segPtr[3]    = { A0, A1, A2 };
    const int             segLen[3]    = { len0, len1, len2 };
    const long long       segStride[3] = { sA0, sA1, sA2 };

    // running B-tile pointers for the two N-tiles (advance 1 tile per K-tile)
    const unsigned int* bp0 = Btiles + ((long long)nt0 * KT) * 256 + lane * 8;
    const unsigned int* bp1 = bp0 + (long long)KT * 256;

    for (int s = 0; s < 3; ++s) {
        int segKT = segLen[s] >> 5;
        if (segKT <= 0) continue;
        long long sA = segStride[s];
        const unsigned short* pa = segPtr[s] + (long long)row * sA + hi * 16;
        const unsigned short* pb = segPtr[s] + (long long)(16 + row) * sA + hi * 16;
        for (int kt = 0; kt < segKT; ++kt) {
            // A fragments: contiguous k = 16*(lane/16) + element, same order as B tiles.
            uint4 a0lo = *(const uint4*)pa;
            uint4 a0hi = *(const uint4*)(pa + 8);
            uint4 a1lo = *(const uint4*)pb;
            uint4 a1hi = *(const uint4*)(pb + 8);
            v8u pk0 = { a0lo.x, a0lo.y, a0lo.z, a0lo.w, a0hi.x, a0hi.y, a0hi.z, a0hi.w };
            v8u pk1 = { a1lo.x, a1lo.y, a1lo.z, a1lo.w, a1hi.x, a1hi.y, a1hi.z, a1hi.w };
            v16bf afrag0 = __builtin_bit_cast(v16bf, pk0);
            v16bf afrag1 = __builtin_bit_cast(v16bf, pk1);

            v8u b0 = *(const v8u*)bp0;
            v8u b1 = *(const v8u*)bp1;
            v16bf bfrag0 = __builtin_bit_cast(v16bf, b0);
            v16bf bfrag1 = __builtin_bit_cast(v16bf, b1);

            acc00 = __builtin_amdgcn_wmma_f32_16x16x32_bf16(false, afrag0, false, bfrag0,
                                                            (short)0, acc00, false, false);
            acc10 = __builtin_amdgcn_wmma_f32_16x16x32_bf16(false, afrag1, false, bfrag0,
                                                            (short)0, acc10, false, false);
            acc01 = __builtin_amdgcn_wmma_f32_16x16x32_bf16(false, afrag0, false, bfrag1,
                                                            (short)0, acc01, false, false);
            acc11 = __builtin_amdgcn_wmma_f32_16x16x32_bf16(false, afrag1, false, bfrag1,
                                                            (short)0, acc11, false, false);

            pa  += 32;
            pb  += 32;
            bp0 += 256;
            bp1 += 256;
        }
    }

    // Epilogue: C/D layout -> row = mt*16 + r + 8*(lane>=16), col = nt*16 + lane%16
    const int colBase = (lane & 15);
    const int rOff    = (lane >> 4) << 3;
    #pragma unroll
    for (int r = 0; r < 8; ++r) {
        int rr0 = r + rOff;         // mt 0
        int rr1 = 16 + r + rOff;    // mt 1
        int c0  = nt0 * 16 + colBase;
        int c1  = (nt0 + 1) * 16 + colBase;
        float v00 = acc00[r], v01 = acc01[r], v10 = acc10[r], v11 = acc11[r];
        if (mode == 1) {
            v00 = tanhf(v00); v01 = tanhf(v01); v10 = tanhf(v10); v11 = tanhf(v11);
            auxbf[(long long)rr0 * ldAux + c0] = f2bf(v00);
            auxbf[(long long)rr0 * ldAux + c1] = f2bf(v01);
            auxbf[(long long)rr1 * ldAux + c0] = f2bf(v10);
            auxbf[(long long)rr1 * ldAux + c1] = f2bf(v11);
        }
        C[(long long)rr0 * ldC + c0] = v00;
        C[(long long)rr0 * ldC + c1] = v01;
        C[(long long)rr1 * ldC + c0] = v10;
        C[(long long)rr1 * ldC + c1] = v11;
    }
}

// ---------------- LSTM cell (elementwise) -----------------------------------

__global__ void lstm_cell(const float* __restrict__ gates,
                          const float* __restrict__ bias,
                          float* __restrict__ c,
                          float* __restrict__ hF,
                          unsigned short* __restrict__ hB) {
    int tid = blockIdx.x * blockDim.x + threadIdx.x;
    if (tid >= BDIM * HDIM) return;
    int b = tid >> 10, n = tid & 1023;
    long long base = (long long)b * 4 * HDIM;
    float gi = gates[base + n]            + bias[n];
    float gf = gates[base + HDIM + n]     + bias[HDIM + n];
    float gg = gates[base + 2 * HDIM + n] + bias[2 * HDIM + n];
    float go = gates[base + 3 * HDIM + n] + bias[3 * HDIM + n];
    float si = 1.f / (1.f + expf(-gi));
    float sf = 1.f / (1.f + expf(-gf));
    float so = 1.f / (1.f + expf(-go));
    float cn = sf * c[tid] + si * tanhf(gg);
    float hn = so * tanhf(cn);
    c[tid]  = cn;
    hF[tid] = hn;
    hB[tid] = f2bf(hn);
}

// ---------------- attention (one block per batch row) -----------------------

__global__ void attn_kernel(const float* __restrict__ q,
                            const float* __restrict__ mem,
                            const unsigned char* __restrict__ mask,
                            float* __restrict__ attn_out, long long ldAttn,
                            unsigned short* __restrict__ ctxb) {
    int b   = blockIdx.x;
    int tid = threadIdx.x;
    __shared__ float qs[HDIM];
    __shared__ float sc[128];
    __shared__ float red[2];
    for (int h = tid; h < HDIM; h += 256) qs[h] = q[b * HDIM + h];
    __syncthreads();
    int w = tid >> 5, lane = tid & 31;
    const float* memb = mem + (long long)b * SDIM * HDIM;
    for (int s = w; s < SDIM; s += 8) {
        const float* m = memb + (long long)s * HDIM;
        float p = 0.f;
        for (int h = lane; h < HDIM; h += 32) p += qs[h] * m[h];
        for (int o = 16; o; o >>= 1) p += __shfl_xor(p, o, 32);
        if (lane == 0) sc[s] = mask[b * SDIM + s] ? p : -1e9f;
    }
    __syncthreads();
    if (w == 0) {
        float mx = -3.4e38f;
        for (int s = lane; s < SDIM; s += 32) mx = fmaxf(mx, sc[s]);
        for (int o = 16; o; o >>= 1) mx = fmaxf(mx, __shfl_xor(mx, o, 32));
        float e = 0.f;
        for (int s = lane; s < SDIM; s += 32) e += expf(sc[s] - mx);
        for (int o = 16; o; o >>= 1) e += __shfl_xor(e, o, 32);
        if (lane == 0) { red[0] = mx; red[1] = e; }
    }
    __syncthreads();
    float mx = red[0], inv = 1.f / red[1];
    for (int s = tid; s < SDIM; s += 256) {
        float a = expf(sc[s] - mx) * inv;
        sc[s] = a;
        attn_out[(long long)b * ldAttn + s] = a;
    }
    __syncthreads();
    for (int h = tid; h < HDIM; h += 256) {
        float acc = 0.f;
        for (int s = 0; s < SDIM; ++s) acc += sc[s] * memb[(long long)s * HDIM + h];
        ctxb[b * HDIM + h] = f2bf(acc);
    }
}

// ---------------- final state copy ------------------------------------------

__global__ void final_copy(const float* __restrict__ h0f, const float* __restrict__ h1f,
                           const float* __restrict__ c0w, const float* __restrict__ c1w,
                           float* __restrict__ outH, float* __restrict__ outC) {
    int tid = blockIdx.x * blockDim.x + threadIdx.x;
    if (tid >= 2 * BDIM * HDIM) return;
    int layer = tid >> 15, idx = tid & 32767;
    outH[tid] = layer ? h1f[idx] : h0f[idx];
    outC[tid] = layer ? c1w[idx] : c0w[idx];
}

// ---------------------------------------------------------------------------

extern "C" void kernel_launch(void* const* d_in, const int* in_sizes, int n_in,
                              void* d_out, int out_size, void* d_ws, size_t ws_size,
                              hipStream_t stream) {
    (void)in_sizes; (void)n_in; (void)out_size; (void)ws_size;
    const float* inputs = (const float*)d_in[0];   // [B,T,E]
    const float* mem    = (const float*)d_in[1];   // [B,S,H]
    const unsigned char* mask = (const unsigned char*)d_in[2]; // [B,S] bool
    const float* h0in   = (const float*)d_in[3];   // [2,B,H]
    const float* c0in   = (const float*)d_in[4];
    const float* W_ih0  = (const float*)d_in[5];   // [4H, E+H]
    const float* W_hh0  = (const float*)d_in[6];   // [4H, H]
    const float* b0     = (const float*)d_in[7];
    const float* W_ih1  = (const float*)d_in[8];   // [4H, H]
    const float* W_hh1  = (const float*)d_in[9];
    const float* b1     = (const float*)d_in[10];
    const float* Wa     = (const float*)d_in[11];  // [H, H]
    const float* Wout   = (const float*)d_in[12];  // [H, 2H]

    float* out = (float*)d_out;
    const long long seqN  = (long long)BDIM * TDIM * HDIM;     // 1,638,400
    const long long attnN = (long long)BDIM * TDIM * SDIM;     //   160,000
    float* outSeq  = out;
    float* outAttn = out + seqN;
    float* outH    = out + seqN + attnN;
    float* outC    = outH + 2 * BDIM * HDIM;

    // ---- workspace carve-up (bytes, 256-aligned) ----
    char* ws = (char*)d_ws;
    size_t off = 0;
    auto carve = [&](size_t bytes) { void* p = ws + off; off += (bytes + 255) & ~(size_t)255; return p; };
    const int K0 = EDIM + 2 * HDIM;   // 2560
    const int K1 = 2 * HDIM;          // 2048
    unsigned int*   wB0  = (unsigned int*)carve((size_t)4 * HDIM * K0 * 2);        // 20.97 MB
    unsigned int*   wB1  = (unsigned int*)carve((size_t)4 * HDIM * K1 * 2);        // 16.78 MB
    unsigned int*   wBa  = (unsigned int*)carve((size_t)HDIM * HDIM * 2);          //  2.10 MB
    unsigned int*   wBo  = (unsigned int*)carve((size_t)HDIM * K1 * 2);            //  4.19 MB
    unsigned short* xbf  = (unsigned short*)carve((size_t)BDIM * TDIM * EDIM * 2); //  1.64 MB
    float*          gates= (float*)carve((size_t)BDIM * 4 * HDIM * 4);
    float*          h0f  = (float*)carve((size_t)BDIM * HDIM * 4);
    float*          h1f  = (float*)carve((size_t)BDIM * HDIM * 4);
    float*          c0w  = (float*)carve((size_t)BDIM * HDIM * 4);
    float*          c1w  = (float*)carve((size_t)BDIM * HDIM * 4);
    unsigned short* h0b  = (unsigned short*)carve((size_t)BDIM * HDIM * 2);
    unsigned short* h1b  = (unsigned short*)carve((size_t)BDIM * HDIM * 2);
    unsigned short* feedb= (unsigned short*)carve((size_t)BDIM * HDIM * 2);
    unsigned short* ctxb = (unsigned short*)carve((size_t)BDIM * HDIM * 2);
    float*          qbuf = (float*)carve((size_t)BDIM * HDIM * 4);

    // ---- one-time conversion & repack ----
    {
        int n = BDIM * TDIM * EDIM;
        cvt_f32_bf16<<<(n + 255) / 256, 256, 0, stream>>>(inputs, xbf, n);
    }
    {
        int nd = (K0 / 2) * 4 * HDIM;     // 5,242,880 dwords
        repack_nk<<<(nd + 255) / 256, 256, 0, stream>>>(wB0, W_ih0, EDIM + HDIM,
                                                        W_hh0, HDIM, K0, 4 * HDIM);
    }
    {
        int nd = (K1 / 2) * 4 * HDIM;
        repack_nk<<<(nd + 255) / 256, 256, 0, stream>>>(wB1, W_ih1, HDIM,
                                                        W_hh1, HDIM, K1, 4 * HDIM);
    }
    {
        int nd = (HDIM / 2) * HDIM;
        repack_kn<<<(nd + 255) / 256, 256, 0, stream>>>(wBa, Wa, HDIM, HDIM);
    }
    {
        int nd = (K1 / 2) * HDIM;
        repack_nk<<<(nd + 255) / 256, 256, 0, stream>>>(wBo, Wout, K1,
                                                        (const float*)nullptr, 0, K1, HDIM);
    }
    {
        int n = 3 * BDIM * HDIM;
        init_state<<<(n + 255) / 256, 256, 0, stream>>>(h0in, c0in, h0f, h1f, c0w, c1w,
                                                        h0b, h1b, feedb);
    }

    // ---- time loop: 7 stream-ordered kernels per step ----
    const int GEMM_THREADS = 128;                    // 4 waves / block
    for (int t = 0; t < TDIM; ++t) {
        // layer 0 gates: A = [x_t | feed | h0], K=2560, N=4096
        {
            int ntGroups = (4 * HDIM) / 32;          // 128 waves
            wmma_gemm32<<<ntGroups / 4, GEMM_THREADS, 0, stream>>>(
                wB0,
                xbf + (long long)t * EDIM, EDIM, (long long)TDIM * EDIM,
                feedb, HDIM, HDIM,
                h0b,   HDIM, HDIM,
                K0, 4 * HDIM, gates, 4 * HDIM, 0, (unsigned short*)nullptr, 0);
        }
        lstm_cell<<<(BDIM * HDIM) / 256, 256, 0, stream>>>(gates, b0, c0w, h0f, h0b);

        // layer 1 gates: A = [h0n | h1], K=2048, N=4096
        {
            int ntGroups = (4 * HDIM) / 32;
            wmma_gemm32<<<ntGroups / 4, GEMM_THREADS, 0, stream>>>(
                wB1,
                h0b, HDIM, HDIM,
                h1b, HDIM, HDIM,
                (const unsigned short*)nullptr, 0, 0,
                K1, 4 * HDIM, gates, 4 * HDIM, 0, (unsigned short*)nullptr, 0);
        }
        lstm_cell<<<(BDIM * HDIM) / 256, 256, 0, stream>>>(gates, b1, c1w, h1f, h1b);

        // q = h1 @ Wa : K=1024, N=1024
        {
            int ntGroups = HDIM / 32;                // 32 waves
            wmma_gemm32<<<ntGroups / 4, GEMM_THREADS, 0, stream>>>(
                wBa,
                h1b, HDIM, HDIM,
                (const unsigned short*)nullptr, 0, 0,
                (const unsigned short*)nullptr, 0, 0,
                HDIM, HDIM, qbuf, HDIM, 0, (unsigned short*)nullptr, 0);
        }

        // attention: scores/softmax/context (writes attn slice + ctx bf16)
        attn_kernel<<<BDIM, 256, 0, stream>>>(qbuf, mem, mask,
                                              outAttn + (long long)t * SDIM,
                                              (long long)TDIM * SDIM, ctxb);

        // o = tanh([ctx | h1] @ Wout^T) -> out seq slice + feed (bf16)
        {
            int ntGroups = HDIM / 32;
            wmma_gemm32<<<ntGroups / 4, GEMM_THREADS, 0, stream>>>(
                wBo,
                ctxb, HDIM, HDIM,
                h1b,  HDIM, HDIM,
                (const unsigned short*)nullptr, 0, 0,
                K1, HDIM, outSeq + (long long)t * HDIM, (long long)TDIM * HDIM,
                1, feedb, HDIM);
        }
    }

    // final hT / cT
    {
        int n = 2 * BDIM * HDIM;
        final_copy<<<(n + 255) / 256, 256, 0, stream>>>(h0f, h1f, c0w, c1w, outH, outC);
    }
}
